// MQALocalAttention_88983132438631
// MI455X (gfx1250) — matrix operand order repaired and checked by reading, not compiled, not run on hardware
//
#include <hip/hip_runtime.h>

// ---------------------------------------------------------------------------
// MQA local attention block for MI455X (gfx1250).
// bf16 WMMA + f32 accumulate; GEMM A-tiles staged to LDS via the Tensor Data
// Mover (tensor_load_to_lds, TENSORcnt), double-buffered against WMMA compute.
// ---------------------------------------------------------------------------

#define DEV __device__ __forceinline__

typedef __bf16        v16bf __attribute__((ext_vector_type(16)));
typedef float         v8f   __attribute__((ext_vector_type(8)));
typedef unsigned int  v8u   __attribute__((ext_vector_type(8)));
typedef unsigned int  u32x4 __attribute__((ext_vector_type(4)));
typedef unsigned int  u32x8 __attribute__((ext_vector_type(8)));

constexpr int Bb = 2;
constexpr int Ss = 2048;
constexpr int Dd = 1024;
constexpr int Hh = 16;
constexpr int HD = 64;
constexpr int BS = Bb * Ss;          // 4096 flattened rows
constexpr float SCALE = 0.125f;      // 1/sqrt(64)
constexpr float NEGV  = -1.0e9f;

DEV unsigned short f2b(float f) {             // f32 -> bf16, round-nearest-even
  unsigned int u = __builtin_bit_cast(unsigned int, f);
  unsigned int r = (u + 0x7FFFu + ((u >> 16) & 1u)) >> 16;
  return (unsigned short)r;
}

// A-fragment: 16x32 bf16 tile at (m0, k0), row-major, contiguous k per lane half (same as frag_b).
// lane L holds row m0 + (L&15); lanes 0-15 K=0..15, lanes 16-31 K=16..31.
DEV v16bf frag_a(const unsigned short* __restrict__ A, int m0, int k0, int lda) {
  const int lane = threadIdx.x & 31;
  const unsigned short* p =
      A + (size_t)(m0 + (lane & 15)) * lda + k0 + ((lane >> 4) << 4);
  v8u t;
#pragma unroll
  for (int i = 0; i < 8; ++i) {
    int k = 2 * i;
    t[i] = *(const unsigned int*)(p + k);
  }
  return __builtin_bit_cast(v16bf, t);
}

// B-fragment from K-transposed storage BT[n][k] (so B(k,n)=BT[n*ldb+k]).
// lane L holds col n0 + (L&15); lanes 0-15 K=0..15, lanes 16-31 K=16..31.
DEV v16bf frag_b(const unsigned short* __restrict__ BT, int n0, int k0, int ldb) {
  const int lane = threadIdx.x & 31;
  const unsigned short* p =
      BT + (size_t)(n0 + (lane & 15)) * ldb + k0 + ((lane >> 4) << 4);
  v8u t;
#pragma unroll
  for (int i = 0; i < 8; ++i) t[i] = *(const unsigned int*)(p + 2 * i);
  return __builtin_bit_cast(v16bf, t);
}

DEV v8f wmma_bf16(v16bf a, v16bf b, v8f c) {
  return __builtin_amdgcn_wmma_f32_16x16x32_bf16(
      /*neg_a=*/false, a, /*neg_b=*/false, b,
      /*c_mod=*/(short)0, c, /*reuse_a=*/false, /*reuse_b=*/false);
}

// ---------------------------------------------------------------------------
// TDM: DMA a 64-row x 32-col bf16 tile (row stride K elements) into LDS.
// D# groups packed per CDNA5 ISA 8.3/8.4; 2D tensor -> groups 2/3 omitted.
// Issued once per wave (EXEC ignored by TDM); tracked by TENSORcnt.
// ---------------------------------------------------------------------------
DEV void tdm_stage_a(const unsigned short* gsrc, unsigned lds_off, unsigned K) {
  unsigned long long ga = (unsigned long long)(size_t)gsrc;
  u32x4 g0;
  g0[0] = 1u;                                        // count=1, user D#
  g0[1] = lds_off;                                   // lds_addr (bytes)
  g0[2] = (unsigned)ga;                              // global_addr[31:0]
  g0[3] = ((unsigned)(ga >> 32) & 0x01FFFFFFu)       // global_addr[56:32]
          | (2u << 30);                              // type=2 ("image")
  const unsigned td0 = K;            // tensor_dim0 (elements)
  const unsigned td1 = 0x100000u;    // tensor_dim1 (large; OOB guard only)
  const unsigned st0 = K;            // tensor_dim0_stride (elements)
  u32x8 g1;
  g1[0] = 1u << 16;                                  // data_size=2B, wg_mask=0
  g1[1] = (td0 & 0xFFFFu) << 16;                     // tensor_dim0[15:0] @63:48
  g1[2] = (td0 >> 16) | ((td1 & 0xFFFFu) << 16);     // td0[31:16] | td1[15:0]
  g1[3] = (td1 >> 16) | (32u << 16);                 // td1[31:16] | tile_dim0=32
  g1[4] = 64u;                                       // tile_dim1=64, tile_dim2=0
  g1[5] = st0;                                       // stride0[31:0]
  g1[6] = 0u;                                        // stride0[47:32], stride1 lo
  g1[7] = 0u;
  asm volatile("tensor_load_to_lds %0, %1" :: "s"(g0), "s"(g1) : "memory");
}

// ---------------------------------------------------------------------------
// Generic bf16 WMMA GEMM: C[M,N] = A[M,K] * B[K,N] + bias.  BT is B stored
// transposed [N][K].  4 waves/block; each wave owns a 64x16 strip.  The 64x32
// A tile shared by all 4 waves is TDM-staged into LDS, double-buffered.
// OUT_MODE: 0 = bf16 row-major, 1 = bf16 transposed (C[n*M+m]), 2 = f32.
// ---------------------------------------------------------------------------
template <int OUT_MODE>
__global__ __launch_bounds__(128) void gemm_bf16_wmma(
    const unsigned short* __restrict__ A, const unsigned short* __restrict__ BT,
    const float* __restrict__ bias, void* __restrict__ C, int M, int N, int K) {
  __shared__ unsigned short As[2][64 * 32];          // 8 KB double buffer
  const int wave = threadIdx.x >> 5;
  const int n0 = blockIdx.x * 64 + wave * 16;
  const int m0 = blockIdx.y * 64;
  v8f acc[4] = {};

  if (wave == 0)   // prologue DMA: first A tile into buffer 0
    tdm_stage_a(A + (size_t)m0 * K, (unsigned)(size_t)(void*)&As[0][0], K);

  for (int k0 = 0; k0 < K; k0 += 32) {
    const int buf = (k0 >> 5) & 1;
    if (wave == 0) __builtin_amdgcn_s_wait_tensorcnt(0);  // current tile landed
    __syncthreads();                                      // visible to all waves
    if (wave == 0 && k0 + 32 < K)                         // overlap next DMA
      tdm_stage_a(A + (size_t)m0 * K + (k0 + 32),
                  (unsigned)(size_t)(void*)&As[buf ^ 1][0], K);

    v16bf b = frag_b(BT, n0, k0, K);
#pragma unroll
    for (int t = 0; t < 4; ++t) {
      v16bf a = frag_a(&As[buf][0], t * 16, 0, 32);       // ds_load path
      acc[t] = wmma_bf16(a, b, acc[t]);
    }
    if (k0 + 64 < K)  // hint next B strip toward the caches (global_prefetch)
      __builtin_prefetch(BT + (size_t)(n0 + (threadIdx.x & 15)) * K + k0 + 64, 0, 0);
  }

  const int lane = threadIdx.x & 31;
  const int half = lane >> 4;
  const int nn = n0 + (lane & 15);
  const float bv = bias ? bias[nn] : 0.0f;
#pragma unroll
  for (int t = 0; t < 4; ++t) {
#pragma unroll
    for (int r = 0; r < 8; ++r) {
      int mm = m0 + t * 16 + r + 8 * half;     // C/D layout: VGPR r -> M=r(+8)
      float val = acc[t][r] + bv;
      if constexpr (OUT_MODE == 0)
        ((unsigned short*)C)[(size_t)mm * N + nn] = f2b(val);
      else if constexpr (OUT_MODE == 1)
        ((unsigned short*)C)[(size_t)nn * M + mm] = f2b(val);
      else
        ((float*)C)[(size_t)mm * N + nn] = val;
    }
  }
}

// ---------------------------------------------------------------------------
// Banded MQA attention: one wave per (batch, head, 16-query tile).
// Window = +/-64 inclusive -> union span 144 keys, padded to 10 tiles (160).
// ---------------------------------------------------------------------------
__global__ __launch_bounds__(32) void attn_kernel(
    const unsigned short* __restrict__ Qb,  // [BS, D]  bf16
    const unsigned short* __restrict__ Kb,  // [BS, 64] bf16
    const unsigned short* __restrict__ Vt,  // [64, BS] bf16 (transposed)
    unsigned short* __restrict__ Ob) {      // [BS, D]  bf16 (pre-Wo)
  __shared__ float scr[16][176];
  __shared__ unsigned short pb[16][176];
  const int q0 = blockIdx.x * 16;
  const int h = blockIdx.y;
  const int b = blockIdx.z;
  const int rb = b * Ss;
  const int lane = threadIdx.x & 31;
  const int half = lane >> 4;

  const v16bf aq0 = frag_a(Qb, rb + q0, h * HD, Dd);
  const v16bf aq1 = frag_a(Qb, rb + q0, h * HD + 32, Dd);

  const int jw0 = q0 - 64;
  for (int jt = 0; jt < 10; ++jt) {
    const int j0 = jw0 + jt * 16;
    // B-fragment of K^T: element(k,n) = K[rb + clamp(j0+n)][k]
    int jn = j0 + (lane & 15);
    int jc = jn < 0 ? 0 : (jn >= Ss ? Ss - 1 : jn);
    const unsigned short* kp = Kb + (size_t)(rb + jc) * HD;
    v8u t0, t1;
#pragma unroll
    for (int i = 0; i < 8; ++i) {
      t0[i] = *(const unsigned int*)(kp + (half << 4) + 2 * i);
      t1[i] = *(const unsigned int*)(kp + 32 + (half << 4) + 2 * i);
    }
    v8f acc = {};
    acc = wmma_bf16(aq0, __builtin_bit_cast(v16bf, t0), acc);
    acc = wmma_bf16(aq1, __builtin_bit_cast(v16bf, t1), acc);
#pragma unroll
    for (int r = 0; r < 8; ++r) {
      int m = r + 8 * half;
      int qi = q0 + m;
      int n = lane & 15;
      int j = j0 + n;
      bool ok = (j >= 0) && (j < Ss) && (j >= qi - 64) && (j < qi + 65);
      scr[m][jt * 16 + n] = ok ? acc[r] * SCALE : NEGV;
    }
  }
  __syncthreads();

  if (lane < 16) {  // one lane per query row: max / exp / sum / pack bf16
    const int m = lane;
    float mx = NEGV;
    for (int c = 0; c < 160; ++c) mx = fmaxf(mx, scr[m][c]);
    float sum = 0.0f;
    for (int c = 0; c < 160; ++c) {
      float e = __expf(scr[m][c] - mx);
      scr[m][c] = e;
      sum += e;
    }
    float inv = 1.0f / sum;
    for (int c = 0; c < 160; ++c) pb[m][c] = f2b(scr[m][c] * inv);
  }
  __syncthreads();

  // P(16x160) @ V(160x64) via 4 n-tiles x 5 k-steps of WMMA.
#pragma unroll
  for (int nt = 0; nt < 4; ++nt) {
    const int n0 = nt * 16;
    v8f acc = {};
#pragma unroll
    for (int kt = 0; kt < 5; ++kt) {
      const int k0 = kt * 32;
      v16bf ap = frag_a(&pb[0][0], 0, k0, 176);
      // B-fragment of V: element(k,n) = Vt[(n0+n)*BS + rb + clamp(jw0+k)]
      const unsigned short* vp = Vt + (size_t)(n0 + (lane & 15)) * BS + rb;
      v8u tv;
#pragma unroll
      for (int i = 0; i < 8; ++i) {
        int jp = jw0 + k0 + (half << 4) + 2 * i;
        jp = jp < 0 ? 0 : (jp > Ss - 2 ? Ss - 2 : jp);  // even clamp; P==0 there
        tv[i] = *(const unsigned int*)(vp + jp);
      }
      acc = wmma_bf16(ap, __builtin_bit_cast(v16bf, tv), acc);
    }
#pragma unroll
    for (int r = 0; r < 8; ++r) {
      int m = r + 8 * half;
      Ob[(size_t)(rb + q0 + m) * Dd + h * HD + n0 + (lane & 15)] = f2b(acc[r]);
    }
  }
}

// ---------------------------------------------------------------------------
// Elementwise / reduction helpers
// ---------------------------------------------------------------------------
__global__ void f32_to_bf16_kernel(const float* __restrict__ src,
                                   unsigned short* __restrict__ dst, int n) {
  int i = blockIdx.x * blockDim.x + threadIdx.x;
  if (i < n) dst[i] = f2b(src[i]);
}

// src: f32 [K][N] -> dst: bf16 [N][K]
__global__ void transpose_bf16_kernel(const float* __restrict__ src,
                                      unsigned short* __restrict__ dst, int K, int N) {
  int i = blockIdx.x * blockDim.x + threadIdx.x;
  if (i < K * N) {
    int k = i / N, n = i % N;
    dst[(size_t)n * K + k] = f2b(src[i]);
  }
}

// concat[s, 0:1024]=bf16(attnO), concat[s, 1024:2048]=bf16(query)
__global__ void pack_concat_kernel(const float* __restrict__ attnO,
                                   const unsigned short* __restrict__ qb16,
                                   unsigned short* __restrict__ cc) {
  int i = blockIdx.x * blockDim.x + threadIdx.x;
  if (i >= BS * 2 * Dd) return;
  int s = i >> 11;
  int c = i & 2047;
  cc[i] = (c < Dd) ? f2b(attnO[(size_t)s * Dd + c])
                   : qb16[(size_t)s * Dd + (c - Dd)];
}

// SiLU gate + gated residual + RMSNorm (per row of D=1024)
__global__ __launch_bounds__(256) void gate_rms_kernel(
    const float* __restrict__ gate_pre, const float* __restrict__ attnO,
    const float* __restrict__ query, const float* __restrict__ rms_w,
    float* __restrict__ out) {
  __shared__ float buf[Dd];
  __shared__ float red[8];
  const int s = blockIdx.x;
  float ss = 0.0f;
  for (int d = threadIdx.x; d < Dd; d += 256) {
    size_t idx = (size_t)s * Dd + d;
    float gp = gate_pre[idx];
    float g = gp / (1.0f + __expf(-gp));           // SiLU
    float o = attnO[idx] * g + query[idx] * (1.0f - g);
    buf[d] = o;
    ss += o * o;
  }
#pragma unroll
  for (int off = 16; off > 0; off >>= 1) ss += __shfl_xor(ss, off, 32);
  if ((threadIdx.x & 31) == 0) red[threadIdx.x >> 5] = ss;
  __syncthreads();
  float tot = 0.0f;
#pragma unroll
  for (int w = 0; w < 8; ++w) tot += red[w];
  float rms = rsqrtf(tot / (float)Dd + 1e-6f);
  for (int d = threadIdx.x; d < Dd; d += 256)
    out[(size_t)s * Dd + d] = buf[d] * rms * rms_w[d];
}

__global__ void colmean_kernel(const float* __restrict__ out,
                               float* __restrict__ mean) {
  int idx = blockIdx.x * blockDim.x + threadIdx.x;
  if (idx >= Bb * Dd) return;
  int b = idx / Dd, d = idx % Dd;
  float s = 0.0f;
  for (int t = 0; t < Ss; ++t) s += out[((size_t)b * Ss + t) * Dd + d];
  mean[idx] = s * (1.0f / (float)Ss);
}

__global__ void submean_kernel(float* __restrict__ out,
                               const float* __restrict__ mean) {
  int i = blockIdx.x * blockDim.x + threadIdx.x;
  if (i >= BS * Dd) return;
  int d = i & (Dd - 1);
  int b = i / (Ss * Dd);
  out[i] -= mean[b * Dd + d];
}

// ---------------------------------------------------------------------------
extern "C" void kernel_launch(void* const* d_in, const int* in_sizes, int n_in,
                              void* d_out, int out_size, void* d_ws, size_t ws_size,
                              hipStream_t stream) {
  (void)in_sizes; (void)n_in; (void)out_size; (void)ws_size;
  const float* query = (const float*)d_in[0];
  const float* key_  = (const float*)d_in[1];
  const float* value = (const float*)d_in[2];
  const float* Wq = (const float*)d_in[3];
  const float* bq = (const float*)d_in[4];
  const float* Wk = (const float*)d_in[5];
  const float* bk = (const float*)d_in[6];
  const float* Wv = (const float*)d_in[7];
  const float* bv = (const float*)d_in[8];
  const float* Wo = (const float*)d_in[9];
  const float* bo = (const float*)d_in[10];
  const float* Wg = (const float*)d_in[11];
  const float* bg = (const float*)d_in[12];
  const float* rms_w = (const float*)d_in[13];
  float* out = (float*)d_out;

  // ---- workspace carve-up (aliased; peak ~65 MB) ----
  char* ws = (char*)d_ws;
  size_t off = 0;
  auto alloc = [&](size_t bytes) -> char* {
    char* p = ws + off;
    off += (bytes + 255) & ~(size_t)255;
    return p;
  };
  unsigned short* xq_b   = (unsigned short*)alloc((size_t)BS * Dd * 2);
  unsigned short* xk_b   = (unsigned short*)alloc((size_t)BS * Dd * 2);
  unsigned short* xv_b   = (unsigned short*)alloc((size_t)BS * Dd * 2);
  unsigned short* q_b    = (unsigned short*)alloc((size_t)BS * Dd * 2);
  unsigned short* attn_b = (unsigned short*)alloc((size_t)BS * Dd * 2);
  unsigned short* wq_t   = (unsigned short*)alloc((size_t)Dd * Dd * 2);
  unsigned short* wk_t   = (unsigned short*)alloc((size_t)HD * Dd * 2);
  unsigned short* wv_t   = (unsigned short*)alloc((size_t)HD * Dd * 2);
  unsigned short* wo_t   = (unsigned short*)alloc((size_t)Dd * Dd * 2);
  unsigned short* wg_t   = (unsigned short*)alloc((size_t)Dd * 2 * Dd * 2);
  unsigned short* k_b    = (unsigned short*)alloc((size_t)BS * HD * 2);
  unsigned short* vt_b   = (unsigned short*)alloc((size_t)HD * BS * 2);
  float* attnO_f = (float*)alloc((size_t)BS * Dd * 4);
  float* cmean   = (float*)alloc((size_t)Bb * Dd * 4);
  unsigned short* concat_b = xk_b;   // reuses xk_b+xv_b (dead after K/V proj)
  float* gate_f = (float*)q_b;       // reuses q_b+attn_b (dead after attn/Wo)

  const int TPB = 256;
  const int nAct = BS * Dd;

  // 1) activations -> bf16
  f32_to_bf16_kernel<<<nAct / TPB, TPB, 0, stream>>>(query, xq_b, nAct);
  f32_to_bf16_kernel<<<nAct / TPB, TPB, 0, stream>>>(key_, xk_b, nAct);
  f32_to_bf16_kernel<<<nAct / TPB, TPB, 0, stream>>>(value, xv_b, nAct);

  // 2) weights -> bf16 transposed [N][K]
  transpose_bf16_kernel<<<(Dd * Dd) / TPB, TPB, 0, stream>>>(Wq, wq_t, Dd, Dd);
  transpose_bf16_kernel<<<(Dd * HD) / TPB, TPB, 0, stream>>>(Wk, wk_t, Dd, HD);
  transpose_bf16_kernel<<<(Dd * HD) / TPB, TPB, 0, stream>>>(Wv, wv_t, Dd, HD);
  transpose_bf16_kernel<<<(Dd * Dd) / TPB, TPB, 0, stream>>>(Wo, wo_t, Dd, Dd);
  transpose_bf16_kernel<<<(2 * Dd * Dd) / TPB, TPB, 0, stream>>>(Wg, wg_t, 2 * Dd, Dd);

  // 3) projections (WMMA, TDM-staged A tiles)
  gemm_bf16_wmma<0><<<dim3(Dd / 64, BS / 64), 128, 0, stream>>>(
      xq_b, wq_t, bq, q_b, BS, Dd, Dd);
  gemm_bf16_wmma<0><<<dim3(1, BS / 64), 128, 0, stream>>>(
      xk_b, wk_t, bk, k_b, BS, HD, Dd);
  gemm_bf16_wmma<1><<<dim3(1, BS / 64), 128, 0, stream>>>(
      xv_b, wv_t, bv, vt_b, BS, HD, Dd);   // V stored transposed [64, BS]

  // 4) banded attention (WMMA scores + WMMA PV)
  attn_kernel<<<dim3(Ss / 16, Hh, Bb), 32, 0, stream>>>(q_b, k_b, vt_b, attn_b);

  // 5) output projection -> f32
  gemm_bf16_wmma<2><<<dim3(Dd / 64, BS / 64), 128, 0, stream>>>(
      attn_b, wo_t, bo, attnO_f, BS, Dd, Dd);

  // 6) concat([attn, query]) -> bf16, then gate GEMM (K=2048) -> f32
  pack_concat_kernel<<<(BS * 2 * Dd) / TPB, TPB, 0, stream>>>(attnO_f, xq_b, concat_b);
  gemm_bf16_wmma<2><<<dim3(Dd / 64, BS / 64), 128, 0, stream>>>(
      concat_b, wg_t, bg, gate_f, BS, Dd, 2 * Dd);

  // 7) SiLU gate + residual + RMSNorm -> d_out
  gate_rms_kernel<<<BS, 256, 0, stream>>>(gate_f, attnO_f, query, rms_w, out);

  // 8) subtract per-(batch,channel) mean over the seq axis
  colmean_kernel<<<(Bb * Dd + TPB - 1) / TPB, TPB, 0, stream>>>(out, cmean);
  submean_kernel<<<(BS * Dd) / TPB, TPB, 0, stream>>>(out, cmean);
}
